// NewtonNet_79577154060800
// MI455X (gfx1250) — matrix operand order repaired and checked by reading, not compiled, not run on hardware
//
#include <hip/hip_runtime.h>

// ---------------------------------------------------------------------------
// NewtonNet forward for MI455X (gfx1250): bf16 WMMA GEMMs, per-atom fusion.
// Weights are pre-transposed/converted once per call into bf16 column-major
// (Wt[n][k], K padded to multiple of 32) so each WMMA B-fragment is one
// contiguous 32-byte load (k = 16g + i per lane half, same k order as the A-fragment).
// msij (64x128 per atom) lives in LDS; cross-atom gathers (a_ms, r_dyn) are
// L2-resident. f_dir / e_dyn are dead w.r.t. the output and are skipped.
// r_dyn / f_dyn use [row][f][3] layout so neighbor gathers are contiguous.
// ---------------------------------------------------------------------------

typedef __bf16 bf16;
typedef __attribute__((ext_vector_type(16))) __bf16 v16bf;
typedef __attribute__((ext_vector_type(8))) float v8f;

#define NF 128
#define RES_ 20
#define AT 256  // atoms per batch (A)

// native hardware converts (v_cvt_*); RNE per CDNA5 VALU rules
__device__ __forceinline__ bf16 f2bf(float f) { return (bf16)f; }
__device__ __forceinline__ float bf2f(bf16 b) { return (float)b; }
__device__ __forceinline__ float silu_f(float x) { return x / (1.f + __expf(-x)); }

__device__ __forceinline__ v8f wmma_bf16(v16bf a, v16bf b, v8f c) {
  // (neg_a, A, neg_b, B, c_mod, C, reuse_a, reuse_b)
  return __builtin_amdgcn_wmma_f32_16x16x32_bf16(false, a, false, b, (short)0, c, false, false);
}

// A fragment (16x32, row-major bf16 tile in LDS, leading dim ldk).
// Lane (g = lane>>4, h = lane&15) holds row h; element i -> k = i+16g
// (contiguous, matching the B-fragment k order). One contiguous 16-elem
// run per lane.
__device__ __forceinline__ v16bf load_a_lds(const bf16* tile, int ldk, int g, int h) {
  v16bf a;
  const bf16* rp = tile + h * ldk;
#pragma unroll
  for (int i = 0; i < 16; ++i) {
    int k = i + 16 * g;
    a[i] = rp[k];
  }
  return a;
}

// B fragment (32x16) from pre-packed bf16 column-major Wt[n][k] (lead dim Kpad).
// Lane holds column n; element i -> k = kt*32 + i + 16g, i.e. one contiguous
// 32-byte (16 x bf16) aligned load per lane.
__device__ __forceinline__ v16bf load_b_bf16(const bf16* __restrict__ Wt, int Kpad, int n, int kt,
                                             int g) {
  return *reinterpret_cast<const v16bf*>(Wt + (size_t)n * Kpad + kt * 32 + 16 * g);
}

// 64x128 = (64x128 bf16 LDS) @ (128x128 bf16 packed) GEMM, result (opt bias,
// opt silu) written bf16 to LDS. 8 waves, wave w owns output cols [16w,16w+16).
template <bool SILU, bool BIAS>
__device__ __forceinline__ void gemm64x128_lds(const bf16* X, const bf16* __restrict__ Wt,
                                               const float* bias, bf16* Y) {
  const int t = threadIdx.x;
  const int lane = t & 31, wave = t >> 5;
  const int g = lane >> 4, h = lane & 15, n0 = wave * 16;
  v16bf bfr[4];
#pragma unroll
  for (int kt = 0; kt < 4; ++kt) bfr[kt] = load_b_bf16(Wt, NF, n0 + h, kt, g);
  float bv = 0.f;
  if constexpr (BIAS) bv = bias[n0 + h];
#pragma unroll
  for (int mt = 0; mt < 4; ++mt) {
    v8f acc = {};
#pragma unroll
    for (int kt = 0; kt < 4; ++kt)
      acc = wmma_bf16(load_a_lds(X + (mt * 16) * NF + kt * 32, NF, g, h), bfr[kt], acc);
#pragma unroll
    for (int v = 0; v < 8; ++v) {
      float x = acc[v] + bv;
      if constexpr (SILU) x = silu_f(x);
      Y[(mt * 16 + v + 8 * g) * NF + n0 + h] = f2bf(x);
    }
  }
}

// ---------------------------------------------------------------------------
// weight prep: Wt[n][k] (bf16, K padded to Kpad with zeros) from f32 W[k][n]
// ---------------------------------------------------------------------------
__global__ __launch_bounds__(256) void prep_w_kernel(const float* __restrict__ W,
                                                     bf16* __restrict__ Wt, int K, int N,
                                                     int Kpad) {
  int idx = blockIdx.x * 256 + threadIdx.x;
  if (idx >= N * Kpad) return;
  int n = idx / Kpad, k = idx % Kpad;
  Wt[idx] = (k < K) ? f2bf(W[k * N + n]) : f2bf(0.f);
}

// ---------------------------------------------------------------------------
// init: a = emb[Z], r_dyn = 0, f_dyn = 0, out = 0
// ---------------------------------------------------------------------------
__global__ __launch_bounds__(256) void init_kernel(const int* __restrict__ Z,
                                                   const float* __restrict__ emb, float* a,
                                                   float* r0, float* fd, float* out) {
  int i = blockIdx.x * blockDim.x + threadIdx.x;
  int stride = gridDim.x * blockDim.x;
  if (i < 4) out[i] = 0.f;
  for (int idx = i; idx < 1024 * NF; idx += stride)
    a[idx] = emb[Z[idx >> 7] * NF + (idx & 127)];
  for (int idx = i; idx < 1024 * 3 * NF; idx += stride) {
    r0[idx] = 0.f;
    fd[idx] = 0.f;
  }
}

// ---------------------------------------------------------------------------
// mlp2: Y = silu(X@W1 + b1) @ W2 + b2 for 1024x128 X. One WG per 64 rows.
// ---------------------------------------------------------------------------
__global__ __launch_bounds__(256) void mlp2_kernel(const float* __restrict__ X,
                                                   const bf16* __restrict__ W1t,
                                                   const float* __restrict__ b1,
                                                   const bf16* __restrict__ W2t,
                                                   const float* __restrict__ b2,
                                                   float* __restrict__ Y) {
  __shared__ bf16 lx[64 * NF];
  __shared__ bf16 lh[64 * NF];
  const int t = threadIdx.x;
  const int row0 = blockIdx.x * 64;
  __builtin_prefetch(W2t + t * 64, 0, 1);  // warm GL2 for second GEMM weights
  for (int i = t; i < 64 * NF; i += 256) lx[i] = f2bf(X[(size_t)row0 * NF + i]);
  __syncthreads();
  gemm64x128_lds<true, true>(lx, W1t, b1, lh);
  __syncthreads();
  const int lane = t & 31, wave = t >> 5, g = lane >> 4, h = lane & 15, n0 = wave * 16;
  v16bf bfr[4];
#pragma unroll
  for (int kt = 0; kt < 4; ++kt) bfr[kt] = load_b_bf16(W2t, NF, n0 + h, kt, g);
  const float bv = b2[n0 + h];
#pragma unroll
  for (int mt = 0; mt < 4; ++mt) {
    v8f acc = {};
#pragma unroll
    for (int kt = 0; kt < 4; ++kt)
      acc = wmma_bf16(load_a_lds(lh + mt * 16 * NF + kt * 32, NF, g, h), bfr[kt], acc);
#pragma unroll
    for (int v = 0; v < 8; ++v)
      Y[(size_t)(row0 + mt * 16 + v + 8 * g) * NF + n0 + h] = acc[v] + bv;
  }
}

// ---------------------------------------------------------------------------
// atom kernel: one workgroup (256 thr, 8 waves) per atom row (b*A+i).
// r_dyn / f_dyn layout: [row][f][3] (d contiguous) for coalesced gathers.
// ---------------------------------------------------------------------------
__global__ __launch_bounds__(256) void atom_kernel(
    const int* __restrict__ Ngl, const float* __restrict__ NMg, const float* __restrict__ Dgl,
    const float* __restrict__ Vgl, const bf16* __restrict__ rbfWt, const float* __restrict__ rbfb,
    const float* __restrict__ fW, const bf16* __restrict__ fsW1t, const float* __restrict__ fsb1,
    const bf16* __restrict__ fsW2t, const float* __restrict__ fsb2,
    const bf16* __restrict__ rxW1t, const bf16* __restrict__ rxW2t,
    const float* __restrict__ ams, const float* __restrict__ rmlp,
    const float* __restrict__ emlp, const float* __restrict__ rold, float* __restrict__ rnew,
    float* __restrict__ fdyn, float* __restrict__ a) {
  __shared__ bf16 lds_msij[64 * NF];  // 16 KB
  __shared__ bf16 lds_h[64 * NF];     // 16 KB
  __shared__ bf16 lds_rbf[64 * 32];   // 4 KB (K padded 20 -> 32)
  __shared__ float lds_ai[NF];
  __shared__ float lds_D[64];
  __shared__ float lds_C[64];
  __shared__ float lds_NM[64];
  __shared__ int lds_N[64];
  __shared__ float lds_fijM[64 * 3];  // masked F_ij = (msij.fW)*NM*V

  const int t = threadIdx.x;
  const int row = blockIdx.x;          // global atom row in [0,1024)
  const int base_b = row & ~(AT - 1);  // batch base row
  const int lane = t & 31, wave = t >> 5;
  const int g = lane >> 4, h = lane & 15, n0 = wave * 16;

  if (t < 64) {
    float d = Dgl[row * 64 + t];
    lds_D[t] = d;
    lds_N[t] = Ngl[row * 64 + t];
    lds_NM[t] = NMg[row * 64 + t];
    // polynomial cutoff, P=9: 1 - 55 x^9 + 99 x^10 - 45 x^11, x = d/5
    float x = d * 0.2f;
    float x2 = x * x;
    float x4 = x2 * x2;
    float x9 = x4 * x4 * x;
    float c = 1.f + x9 * (-55.f + 99.f * x - 45.f * x2);
    lds_C[t] = (x < 1.f) ? c : 0.f;
  }
  if (t < NF) lds_ai[t] = ams[(size_t)row * NF + t];
  for (int i = t; i < 64 * 32; i += 256) lds_rbf[i] = f2bf(0.f);
  __syncthreads();
  // Bessel RBF: sqrt(2/5) * sin(n*pi*d/5) / d, n = 1..20
  for (int i = t; i < 64 * RES_; i += 256) {
    int nn = i / RES_, r = i % RES_;
    float d = lds_D[nn];
    float val = 0.6324555320336759f * __sinf((float)(r + 1) * 0.6283185307179586f * d) / d;
    lds_rbf[nn * 32 + r] = f2bf(val);
  }
  __syncthreads();

  // rbf_ms = (rbf @ rbfW + rbfb) * C; msij = rbf_ms * a_ms[j] * a_ms[i]
  {
    v16bf bfr = load_b_bf16(rbfWt, 32, n0 + h, 0, g);
    const float bv = rbfb[n0 + h];
#pragma unroll
    for (int mt = 0; mt < 4; ++mt) {
      v8f acc = {};
      acc = wmma_bf16(load_a_lds(lds_rbf + mt * 16 * 32, 32, g, h), bfr, acc);
#pragma unroll
      for (int v = 0; v < 8; ++v) {
        int nn = mt * 16 + v + 8 * g;
        int f = n0 + h;
        float rm = (acc[v] + bv) * lds_C[nn];
        float aj = ams[(size_t)(base_b + lds_N[nn]) * NF + f];
        lds_msij[nn * NF + f] = f2bf(rm * aj * lds_ai[f]);
      }
    }
  }
  __syncthreads();

  // Masked directional message: fijM[nn][d] = (msij[nn,:].fW) * NM[nn] * V[nn][d]
  if (t < 64) {
    float dot = 0.f;
#pragma unroll 4
    for (int f = 0; f < NF; ++f) dot += bf2f(lds_msij[t * NF + f]) * fW[f];
    dot *= lds_NM[t];
#pragma unroll
    for (int d = 0; d < 3; ++d)
      lds_fijM[t * 3 + d] = dot * Vgl[((size_t)row * 64 + t) * 3 + d];
  }
  __syncthreads();

  // fs = silu(msij@fsW1+b1)@fsW2+b2; F_i[d][f] = sum_nn fs * fijM  (fused)
  gemm64x128_lds<true, true>(lds_msij, fsW1t, fsb1, lds_h);
  __syncthreads();
  float fi[3] = {0.f, 0.f, 0.f};
  {
    v16bf bfr[4];
#pragma unroll
    for (int kt = 0; kt < 4; ++kt) bfr[kt] = load_b_bf16(fsW2t, NF, n0 + h, kt, g);
    const float bv = fsb2[n0 + h];
#pragma unroll
    for (int mt = 0; mt < 4; ++mt) {
      v8f acc = {};
#pragma unroll
      for (int kt = 0; kt < 4; ++kt)
        acc = wmma_bf16(load_a_lds(lds_h + mt * 16 * NF + kt * 32, NF, g, h), bfr[kt], acc);
#pragma unroll
      for (int v = 0; v < 8; ++v) {
        int nn = mt * 16 + v + 8 * g;
        float fsv = acc[v] + bv;
        fi[0] += fsv * lds_fijM[nn * 3 + 0];
        fi[1] += fsv * lds_fijM[nn * 3 + 1];
        fi[2] += fsv * lds_fijM[nn * 3 + 2];
      }
    }
#pragma unroll
    for (int d = 0; d < 3; ++d) fi[d] += __shfl_xor(fi[d], 16, 32);  // merge row halves
  }
  __syncthreads();

  // rext = silu(msij@rxW1)@rxW2 (no bias); dr_ext[d][f] = sum_nn NM*rext*r_old[j]
  gemm64x128_lds<true, false>(lds_msij, rxW1t, nullptr, lds_h);
  __syncthreads();
  float dre[3] = {0.f, 0.f, 0.f};
  {
    v16bf bfr[4];
#pragma unroll
    for (int kt = 0; kt < 4; ++kt) bfr[kt] = load_b_bf16(rxW2t, NF, n0 + h, kt, g);
    const int f = n0 + h;
#pragma unroll
    for (int mt = 0; mt < 4; ++mt) {
      v8f acc = {};
#pragma unroll
      for (int kt = 0; kt < 4; ++kt)
        acc = wmma_bf16(load_a_lds(lds_h + mt * 16 * NF + kt * 32, NF, g, h), bfr[kt], acc);
#pragma unroll
      for (int v = 0; v < 8; ++v) {
        int nn = mt * 16 + v + 8 * g;
        float w = acc[v] * lds_NM[nn];
        // [row][f][3] layout: 3 contiguous floats per (neighbor, f)
        const float* rp = rold + ((size_t)(base_b + lds_N[nn]) * NF + f) * 3;
        dre[0] += w * rp[0];
        dre[1] += w * rp[1];
        dre[2] += w * rp[2];
      }
    }
#pragma unroll
    for (int d = 0; d < 3; ++d) dre[d] += __shfl_xor(dre[d], 16, 32);
  }

  // State update for this atom (column f = n0+h, g==0 lanes hold merged sums)
  if (g == 0) {
    const int f = n0 + h;
    float rmv = rmlp[(size_t)row * NF + f];
    float rn[3], fdn[3];
    const size_t ix = ((size_t)row * NF + f) * 3;
#pragma unroll
    for (int d = 0; d < 3; ++d) {
      float ro = rold[ix + d];
      rn[d] = ro + rmv * fi[d] + dre[d];  // r_dyn += dr_i + dr_ext
      rnew[ix + d] = rn[d];
      fdn[d] = fdyn[ix + d] + fi[d];      // f_dyn += F_i
      fdyn[ix + d] = fdn[d];
    }
    float de = -(fdn[0] * rn[0] + fdn[1] * rn[1] + fdn[2] * rn[2]);
    de *= emlp[(size_t)row * NF + f];
    a[(size_t)row * NF + f] += de;        // a += de_i
  }
}

// ---------------------------------------------------------------------------
// head: silu(a@W1+b1) -> silu(@W2+b2) -> @W3+b3 -> masked per-batch sum
// ---------------------------------------------------------------------------
__global__ __launch_bounds__(256) void head_kernel(const float* __restrict__ a,
                                                   const bf16* __restrict__ W1t,
                                                   const float* __restrict__ b1,
                                                   const bf16* __restrict__ W2t,
                                                   const float* __restrict__ b2,
                                                   const float* __restrict__ W3,
                                                   const float* __restrict__ b3,
                                                   const float* __restrict__ AM,
                                                   float* __restrict__ out) {
  __shared__ bf16 lx[64 * NF];
  __shared__ bf16 lh[64 * NF];
  const int t = threadIdx.x;
  const int row0 = blockIdx.x * 64;
  for (int i = t; i < 64 * NF; i += 256) lx[i] = f2bf(a[(size_t)row0 * NF + i]);
  __syncthreads();
  gemm64x128_lds<true, true>(lx, W1t, b1, lh);
  __syncthreads();
  const int lane = t & 31, wave = t >> 5, g = lane >> 4, h = lane & 15;
  if (wave < 4) {  // N = 64 -> 4 column tiles; wave-uniform branch (EXEC all-1 inside)
    const int n0 = wave * 16;
    v16bf bfr[4];
#pragma unroll
    for (int kt = 0; kt < 4; ++kt) bfr[kt] = load_b_bf16(W2t, NF, n0 + h, kt, g);
    const float bv = b2[n0 + h];
#pragma unroll
    for (int mt = 0; mt < 4; ++mt) {
      v8f acc = {};
#pragma unroll
      for (int kt = 0; kt < 4; ++kt)
        acc = wmma_bf16(load_a_lds(lh + mt * 16 * NF + kt * 32, NF, g, h), bfr[kt], acc);
#pragma unroll
      for (int v = 0; v < 8; ++v)
        lx[(mt * 16 + v + 8 * g) * 64 + n0 + h] = f2bf(silu_f(acc[v] + bv));
    }
  }
  __syncthreads();
  if (t < 64) {
    float s = 0.f;
#pragma unroll 4
    for (int f = 0; f < 64; ++f) s += bf2f(lx[t * 64 + f]) * W3[f];
    s += b3[0];
    int grow = row0 + t;
    s *= AM[grow];
    atomicAdd(&out[grow >> 8], s);  // per-batch masked sum
  }
}

// ---------------------------------------------------------------------------
// host launcher
// ---------------------------------------------------------------------------
extern "C" void kernel_launch(void* const* d_in, const int* in_sizes, int n_in, void* d_out,
                              int out_size, void* d_ws, size_t ws_size, hipStream_t stream) {
  (void)in_sizes;
  (void)n_in;
  (void)out_size;
  (void)ws_size;
  // setup_inputs() dict order (params dict flattened in insertion order)
  const int* Z = (const int*)d_in[0];
  const int* N = (const int*)d_in[1];
  const float* NM = (const float*)d_in[2];
  const float* AM = (const float*)d_in[3];
  const float* D = (const float*)d_in[4];
  const float* V = (const float*)d_in[5];
  const float* emb = (const float*)d_in[6];
  const float* rbfW = (const float*)d_in[7];
  const float* rbfb = (const float*)d_in[8];
  const float* aW1 = (const float*)d_in[9];
  const float* ab1 = (const float*)d_in[10];
  const float* aW2 = (const float*)d_in[11];
  const float* ab2 = (const float*)d_in[12];
  const float* fWt = (const float*)d_in[13];
  const float* fsW1 = (const float*)d_in[14];
  const float* fsb1 = (const float*)d_in[15];
  const float* fsW2 = (const float*)d_in[16];
  const float* fsb2 = (const float*)d_in[17];
  const float* rW1 = (const float*)d_in[18];
  const float* rb1 = (const float*)d_in[19];
  const float* rW2 = (const float*)d_in[20];
  const float* rb2 = (const float*)d_in[21];
  const float* rxW1 = (const float*)d_in[22];
  const float* rxW2 = (const float*)d_in[23];
  const float* eW1 = (const float*)d_in[24];
  const float* eb1 = (const float*)d_in[25];
  const float* eW2 = (const float*)d_in[26];
  const float* eb2 = (const float*)d_in[27];
  const float* aeW1 = (const float*)d_in[28];
  const float* aeb1 = (const float*)d_in[29];
  const float* aeW2 = (const float*)d_in[30];
  const float* aeb2 = (const float*)d_in[31];
  const float* aeW3 = (const float*)d_in[32];
  const float* aeb3 = (const float*)d_in[33];

  const int ROWS = 1024;
  float* ws = (float*)d_ws;
  float* a = ws;      ws += ROWS * NF;
  float* ams = ws;    ws += ROWS * NF;
  float* rmlp = ws;   ws += ROWS * NF;
  float* emlp = ws;   ws += ROWS * NF;
  float* r0 = ws;     ws += ROWS * 3 * NF;
  float* r1 = ws;     ws += ROWS * 3 * NF;
  float* fd = ws;     ws += ROWS * 3 * NF;
  float* out = (float*)d_out;

  // bf16 packed-weight region (32B aligned: float region is 6815744 bytes)
  bf16* wb = (bf16*)ws;
  const size_t SQ = (size_t)NF * NF;           // 16384
  const size_t LBF = 10 * SQ + 32 * NF;        // per-layer bf16 block
  auto lw = [&](int l, int slot) { return wb + (size_t)l * LBF + (size_t)slot * SQ; };
  bf16* headW1t = wb + 3 * LBF;
  bf16* headW2t = headW1t + SQ;

  auto PREP = [&](const float* W, bf16* Wt, int K, int Nn, int Kpad) {
    int total = Nn * Kpad;
    prep_w_kernel<<<(total + 255) / 256, 256, 0, stream>>>(W, Wt, K, Nn, Kpad);
  };

  for (int l = 0; l < 3; ++l) {
    const int wOff = l * NF * NF;
    PREP(aW1 + wOff, lw(l, 0), NF, NF, NF);
    PREP(aW2 + wOff, lw(l, 1), NF, NF, NF);
    PREP(rW1 + wOff, lw(l, 2), NF, NF, NF);
    PREP(rW2 + wOff, lw(l, 3), NF, NF, NF);
    PREP(eW1 + wOff, lw(l, 4), NF, NF, NF);
    PREP(eW2 + wOff, lw(l, 5), NF, NF, NF);
    PREP(fsW1 + wOff, lw(l, 6), NF, NF, NF);
    PREP(fsW2 + wOff, lw(l, 7), NF, NF, NF);
    PREP(rxW1 + wOff, lw(l, 8), NF, NF, NF);
    PREP(rxW2 + wOff, lw(l, 9), NF, NF, NF);
    PREP(rbfW + l * RES_ * NF, lw(l, 10), RES_, NF, 32);  // 20 -> 32 zero pad
  }
  PREP(aeW1, headW1t, NF, NF, NF);
  PREP(aeW2, headW2t, NF, 64, NF);

  init_kernel<<<512, 256, 0, stream>>>(Z, emb, a, r0, fd, out);

  for (int l = 0; l < 3; ++l) {
    const int bOff = l * NF;
    mlp2_kernel<<<16, 256, 0, stream>>>(a, lw(l, 0), ab1 + bOff, lw(l, 1), ab2 + bOff, ams);
    mlp2_kernel<<<16, 256, 0, stream>>>(a, lw(l, 2), rb1 + bOff, lw(l, 3), rb2 + bOff, rmlp);
    mlp2_kernel<<<16, 256, 0, stream>>>(a, lw(l, 4), eb1 + bOff, lw(l, 5), eb2 + bOff, emlp);
    const float* rold = (l & 1) ? r1 : r0;
    float* rnew = (l & 1) ? r0 : r1;
    atom_kernel<<<1024, 256, 0, stream>>>(N, NM, D, V, lw(l, 10), rbfb + bOff, fWt + bOff,
                                          lw(l, 6), fsb1 + bOff, lw(l, 7), fsb2 + bOff, lw(l, 8),
                                          lw(l, 9), ams, rmlp, emlp, rold, rnew, fd, a);
  }

  head_kernel<<<16, 256, 0, stream>>>(a, headW1t, aeb1, headW2t, aeb2, aeW3, aeb3, AM, out);
}
